// Attention_335007449342
// MI455X (gfx1250) — matrix operand order repaired and checked by reading, not compiled, not run on hardware
//
#include <hip/hip_runtime.h>
#include <hip/hip_bf16.h>

#define B_ 2
#define S_ 2048
#define D_ 4096
#define H_ 32
#define HD_ 128
#define AL_ 10
#define R_ 16

#if __has_builtin(__builtin_amdgcn_global_load_async_to_lds_b128)
#define HAVE_ASYNC 1
#else
#define HAVE_ASYNC 0
#endif

typedef __attribute__((ext_vector_type(16))) __bf16 v16bf;
typedef __attribute__((ext_vector_type(8)))  float  v8f;
typedef __attribute__((__vector_size__(4 * sizeof(int)))) int i32x4_;
typedef __attribute__((address_space(1))) i32x4_* gas_i32x4p;
typedef __attribute__((address_space(3))) i32x4_* las_i32x4p;

union FragH { v16bf v; uint4 u[2]; };

static __device__ __forceinline__ v8f wmma_bf16(const FragH& a, const FragH& b, v8f c) {
  // D = A(16x32 bf16) * B(32x16 bf16) + C(16x16 f32)
  return __builtin_amdgcn_wmma_f32_16x16x32_bf16(false, a.v, false, b.v, (short)0, c,
                                                 false, false);
}

static __device__ __forceinline__ float rowmax16(float v) {
#pragma unroll
  for (int off = 1; off < 16; off <<= 1) v = fmaxf(v, __shfl_xor(v, off, 32));
  return v;
}
static __device__ __forceinline__ float rowsum16(float v) {
#pragma unroll
  for (int off = 1; off < 16; off <<= 1) v += __shfl_xor(v, off, 32);
  return v;
}

// 16B global -> LDS copy: async CDNA5 path (ASYNCcnt) if available.
static __device__ __forceinline__ void cp_g2l_16B(const __bf16* g, __bf16* l) {
#if HAVE_ASYNC
  __builtin_amdgcn_global_load_async_to_lds_b128((gas_i32x4p)(i32x4_*)g,
                                                 (las_i32x4p)(i32x4_*)l, 0, 0);
#else
  *(uint4*)l = *(const uint4*)g;
#endif
}

static __device__ __forceinline__ void wait_async_done() {
#if HAVE_ASYNC
#if __has_builtin(__builtin_amdgcn_s_wait_asynccnt)
  __builtin_amdgcn_s_wait_asynccnt(0);
#else
  asm volatile("s_wait_asynccnt 0" ::: "memory");
#endif
#endif
}

// ---------------------------------------------------------------------------
// Elementwise f32 -> bf16
// ---------------------------------------------------------------------------
__global__ void cvt_f32_bf16(const float* __restrict__ in, __bf16* __restrict__ out,
                             size_t n) {
  size_t i = (size_t)blockIdx.x * blockDim.x + threadIdx.x;
  if (i < n) out[i] = (__bf16)in[i];
}

// ---------------------------------------------------------------------------
// Transpose xv (B,S,D) -> vT (B,H,HD,S), bf16
// ---------------------------------------------------------------------------
__global__ void transpose_v_kernel(const __bf16* __restrict__ xv,
                                   __bf16* __restrict__ vT) {
  size_t i = (size_t)blockIdx.x * blockDim.x + threadIdx.x;
  const size_t total = (size_t)B_ * H_ * HD_ * S_;
  if (i >= total) return;
  int s  = (int)(i % S_);
  size_t j = i / S_;
  int hd = (int)(j % HD_); j /= HD_;
  int h  = (int)(j % H_);
  int b  = (int)(j / H_);
  vT[i] = xv[((size_t)(b * S_ + s)) * D_ + h * HD_ + hd];
}

// ---------------------------------------------------------------------------
// av (B,AL,D) -> avT (B,H,HD,32) bf16, zero padded keys [AL,32)
// ---------------------------------------------------------------------------
__global__ void make_avT_kernel(const __bf16* __restrict__ av,
                                __bf16* __restrict__ avT) {
  int i = blockIdx.x * blockDim.x + threadIdx.x;
  const int total = B_ * H_ * HD_ * 32;
  if (i >= total) return;
  int k = i & 31;
  int j = i >> 5;
  int hd = j & (HD_ - 1); j >>= 7;
  int h  = j & (H_ - 1);
  int b  = j >> 5;
  avT[i] = (k < AL_) ? av[((size_t)(b * AL_ + k)) * D_ + h * HD_ + hd] : (__bf16)0.f;
}

// ---------------------------------------------------------------------------
// Cooperative GEMM for M,N multiples of 128:
//   out(MxN) = A(MxK,bf16) @ W(NxK,bf16)^T [+ tlo(Mx16) @ l2(Nx16)^T]
// Block = 8 waves = 128x128 tile. W k-slice (128x32) double-buffered in LDS
// via async global->LDS loads; all 8 waves share it (8x less W HBM traffic).
// ---------------------------------------------------------------------------
__global__ __launch_bounds__(256) void gemm_bf16_coop(
    const __bf16* __restrict__ A, const __bf16* __restrict__ W,
    float* __restrict__ outF, __bf16* __restrict__ outB,
    const float* __restrict__ tlo, const float* __restrict__ l2,
    int M, int N, int K, int nblk) {
  __shared__ __bf16 wtile[2][128 * 32];
  const int tid  = threadIdx.x;
  const int lane = tid & 31;
  const int wsl  = __builtin_amdgcn_readfirstlane(tid >> 5);
  const int bm   = blockIdx.x / nblk;
  const int bn   = blockIdx.x % nblk;
  const int m0   = bm * 128 + wsl * 16;
  const int n0b  = bn * 128;
  const int grp  = lane >> 4;
  const int l16  = lane & 15;
  const int ka   = grp << 4;   // A fragment K chunk base
  const int kb   = grp << 4;   // B fragment K base

  v8f zero8 = {};
  v8f acc[8];
#pragma unroll
  for (int j = 0; j < 8; ++j) acc[j] = zero8;

  const __bf16* arow = A + (size_t)(m0 + l16) * K;

  // Prologue: stage k=0 W tile (rows n0b..n0b+127, cols 0..31)
#pragma unroll
  for (int i = 0; i < 2; ++i) {
    const int idx = tid * 2 + i;        // 512 16B-chunks
    const int row = idx >> 2;
    const int ch  = (idx & 3) << 3;
    cp_g2l_16B(W + (size_t)(n0b + row) * K + ch, &wtile[0][row * 32 + ch]);
  }
  wait_async_done();
  __syncthreads();

  int cur = 0;
  for (int k = 0; k < K; k += 32) {
    if (k + 32 < K) {  // uniform: stage next k-slice while computing this one
#pragma unroll
      for (int i = 0; i < 2; ++i) {
        const int idx = tid * 2 + i;
        const int row = idx >> 2;
        const int ch  = (idx & 3) << 3;
        cp_g2l_16B(W + (size_t)(n0b + row) * K + (k + 32) + ch,
                   &wtile[cur ^ 1][row * 32 + ch]);
      }
    }
    FragH af;
    const uint4* pa = (const uint4*)(arow + k + ka);
    af.u[0] = pa[0];
    af.u[1] = pa[1];
    FragH bf[8];
#pragma unroll
    for (int j = 0; j < 8; ++j) {  // all B frags first -> one wait, 8 WMMAs
      const __bf16* lr = &wtile[cur][(j * 16 + l16) * 32 + kb];
      bf[j].u[0] = *(const uint4*)lr;
      bf[j].u[1] = *(const uint4*)(lr + 8);
    }
#pragma unroll
    for (int j = 0; j < 8; ++j) acc[j] = wmma_bf16(af, bf[j], acc[j]);
    wait_async_done();
    __syncthreads();
    cur ^= 1;
  }

  const int mr0 = m0 + (grp << 3);

  if (tlo != nullptr) {  // fused LoRA epilogue
#pragma unroll
    for (int j = 0; j < 8; ++j) {
      const float* l2r = l2 + (size_t)(n0b + (j << 4) + l16) * R_;
      float lv[R_];
#pragma unroll
      for (int c = 0; c < R_; ++c) lv[c] = l2r[c];
#pragma unroll
      for (int r = 0; r < 8; ++r) {
        const float* tr = tlo + (size_t)(mr0 + r) * R_;
        float s = 0.f;
#pragma unroll
        for (int c = 0; c < R_; ++c) s += tr[c] * lv[c];
        acc[j][r] += s;
      }
    }
  }

#pragma unroll
  for (int j = 0; j < 8; ++j) {
    const int n = n0b + (j << 4) + l16;
#pragma unroll
    for (int r = 0; r < 8; ++r) {
      const size_t o = (size_t)(mr0 + r) * N + n;
      if (outF) outF[o] = acc[j][r];
      else      outB[o] = (__bf16)acc[j][r];
    }
  }
}

// ---------------------------------------------------------------------------
// Simple per-wave GEMM for the small shapes (N=16 LoRA-t, M=20 adapter).
// Wave id scalarized so tile guards are SALU branches (EXEC untouched).
// ---------------------------------------------------------------------------
__global__ __launch_bounds__(256) void gemm_bf16_wmma(
    const __bf16* __restrict__ A, const __bf16* __restrict__ W,
    float* __restrict__ outF, __bf16* __restrict__ outB,
    int M, int N, int K) {
  const int lane = threadIdx.x & 31;
  const int wave = __builtin_amdgcn_readfirstlane(
      blockIdx.x * (blockDim.x >> 5) + (threadIdx.x >> 5));
  const int tilesM = (M + 15) >> 4;
  const int tilesN = (N + 127) >> 7;
  if (wave >= tilesM * tilesN) return;  // scalar branch
  const int tm = wave / tilesN, tn = wave % tilesN;
  const int m0 = tm << 4, n0b = tn << 7;
  const int grp = lane >> 4;
  const int l16 = lane & 15;
  int row_a = m0 + l16; if (row_a > M - 1) row_a = M - 1;
  const int ka = grp << 4;
  const int kb = grp << 4;
  int nsub = (N - n0b) >> 4; if (nsub > 8) nsub = 8;  // scalar

  v8f zero8 = {};
  v8f acc[8];
#pragma unroll
  for (int j = 0; j < 8; ++j) acc[j] = zero8;

  const __bf16* arow = A + (size_t)row_a * K;
  for (int k = 0; k < K; k += 32) {
    FragH af;
    const uint4* pa = (const uint4*)(arow + k + ka);
    af.u[0] = pa[0];
    af.u[1] = pa[1];
#pragma unroll
    for (int j = 0; j < 8; ++j) {
      if (j < nsub) {  // scalar compare -> s_cbranch
        const __bf16* brow = W + (size_t)(n0b + (j << 4) + l16) * K;
        FragH bf_;
        const uint4* pb = (const uint4*)(brow + k + kb);
        bf_.u[0] = pb[0];
        bf_.u[1] = pb[1];
        acc[j] = wmma_bf16(af, bf_, acc[j]);
      }
    }
  }

  const int mr0 = m0 + (grp << 3);
#pragma unroll
  for (int j = 0; j < 8; ++j) {
    if (j < nsub) {
      const int n = n0b + (j << 4) + l16;
#pragma unroll
      for (int r = 0; r < 8; ++r) {
        const int m = mr0 + r;
        if (m < M) {
          const size_t o = (size_t)m * N + n;
          if (outF) outF[o] = acc[j][r];
          else      outB[o] = (__bf16)acc[j][r];
        }
      }
    }
  }
}

// ---------------------------------------------------------------------------
// Flash attention + gated adapter cross-attention.
// One wave per (b, h, 16-query tile). Q resident; KV in blocks of 32.
// ---------------------------------------------------------------------------
__global__ __launch_bounds__(256) void flash_attn_wmma(
    const __bf16* __restrict__ xq, const __bf16* __restrict__ xk,
    const __bf16* __restrict__ vT, const __bf16* __restrict__ ak,
    const __bf16* __restrict__ avT, const float* __restrict__ gate,
    const float* __restrict__ ngate, __bf16* __restrict__ out) {
  __shared__ float pbuf[8][16][33];  // per-wave P exchange tile
  const int lane = threadIdx.x & 31;
  const int wsl  = __builtin_amdgcn_readfirstlane(threadIdx.x >> 5);
  const int wave = __builtin_amdgcn_readfirstlane(blockIdx.x * 8 + wsl);
  const int qt = wave & (S_ / 16 - 1);
  const int h  = (wave >> 7) & (H_ - 1);
  const int b  = wave >> 12;
  const int q0 = qt << 4;
  const int l16 = lane & 15;
  const int grp = lane >> 4;
  const int ka = grp << 4;
  const int kb = grp << 4;
  const int rb = grp << 3;
  const float scale = 0.08838834764831845f;  // 1/sqrt(128)

  FragH qf[4];
  {
    const __bf16* qbase = xq + ((size_t)(b * S_ + q0 + l16)) * D_ + h * HD_;
#pragma unroll
    for (int kk = 0; kk < 4; ++kk) {
      const uint4* pq = (const uint4*)(qbase + kk * 32 + ka);
      qf[kk].u[0] = pq[0];
      qf[kk].u[1] = pq[1];
    }
  }

  v8f zero8 = {};
  v8f acc[8];
#pragma unroll
  for (int j = 0; j < 8; ++j) acc[j] = zero8;
  float mm[8], ll[8];
#pragma unroll
  for (int r = 0; r < 8; ++r) { mm[r] = -1e30f; ll[r] = 0.f; }

  const __bf16* vbase = vT + (size_t)((b * H_ + h) * HD_) * S_;

  for (int kv0 = 0; kv0 < q0 + 16; kv0 += 32) {
    v8f sc[2];
#pragma unroll
    for (int nt = 0; nt < 2; ++nt) {
      sc[nt] = zero8;
      const __bf16* kbase2 =
          xk + ((size_t)(b * S_ + kv0 + nt * 16 + l16)) * D_ + h * HD_;
#pragma unroll
      for (int kk = 0; kk < 4; ++kk) {
        FragH bf_;
        const uint4* pb = (const uint4*)(kbase2 + kk * 32 + kb);
        bf_.u[0] = pb[0];
        bf_.u[1] = pb[1];
        sc[nt] = wmma_bf16(qf[kk], bf_, sc[nt]);
      }
    }
#pragma unroll
    for (int r = 0; r < 8; ++r) {
      const int qg = q0 + rb + r;
      float rmax = -1e30f;
#pragma unroll
      for (int nt = 0; nt < 2; ++nt) {
        const int key = kv0 + nt * 16 + l16;
        float s = sc[nt][r] * scale + (key > qg ? -1e9f : 0.f);
        sc[nt][r] = s;
        rmax = fmaxf(rmax, s);
      }
      rmax = rowmax16(rmax);
      const float mnew = fmaxf(mm[r], rmax);
      const float corr = __expf(mm[r] - mnew);
      float rsum = 0.f;
#pragma unroll
      for (int nt = 0; nt < 2; ++nt) {
        const float pe = __expf(sc[nt][r] - mnew);
        sc[nt][r] = pe;
        rsum += pe;
      }
      rsum = rowsum16(rsum);
      ll[r] = ll[r] * corr + rsum;
      mm[r] = mnew;
#pragma unroll
      for (int j = 0; j < 8; ++j) acc[j][r] *= corr;
      pbuf[wsl][rb + r][l16]      = sc[0][r];
      pbuf[wsl][rb + r][16 + l16] = sc[1][r];
    }
    asm volatile("s_wait_dscnt 0" ::: "memory");
    FragH pf;
#pragma unroll
    for (int i = 0; i < 8; ++i) {
      pf.v[i]     = (__bf16)pbuf[wsl][l16][ka + i];
      pf.v[8 + i] = (__bf16)pbuf[wsl][l16][ka + 8 + i];
    }
#pragma unroll
    for (int j = 0; j < 8; ++j) {
      FragH vf;
      const uint4* pv = (const uint4*)(vbase + (size_t)(j * 16 + l16) * S_ + kv0 + kb);
      vf.u[0] = pv[0];
      vf.u[1] = pv[1];
      acc[j] = wmma_bf16(pf, vf, acc[j]);
    }
  }

#pragma unroll
  for (int r = 0; r < 8; ++r) {
    const float inv = 1.f / ll[r];
#pragma unroll
    for (int j = 0; j < 8; ++j) acc[j][r] *= inv;
  }

  // gated adapter cross-attention (AL=10 keys, padded to 16/32)
  {
    v8f asc = zero8;
    const int keyc = l16 < AL_ ? l16 : AL_ - 1;
    const __bf16* akbase = ak + ((size_t)(b * AL_ + keyc)) * D_ + h * HD_;
#pragma unroll
    for (int kk = 0; kk < 4; ++kk) {
      FragH bf_;
      const uint4* pb = (const uint4*)(akbase + kk * 32 + kb);
      bf_.u[0] = pb[0];
      bf_.u[1] = pb[1];
      asc = wmma_bf16(qf[kk], bf_, asc);
    }
    const float g = tanhf(gate[h]) * ngate[0];
#pragma unroll
    for (int r = 0; r < 8; ++r) {
      float s = (l16 < AL_) ? asc[r] * scale : -1e30f;
      float rmax = rowmax16(s);
      float pe = __expf(s - rmax);
      float rsum = rowsum16(pe);
      float ap = g * pe / rsum;
      pbuf[wsl][rb + r][l16]      = ap;
      pbuf[wsl][rb + r][16 + l16] = 0.f;
    }
    asm volatile("s_wait_dscnt 0" ::: "memory");
    FragH pf;
#pragma unroll
    for (int i = 0; i < 8; ++i) {
      pf.v[i]     = (__bf16)pbuf[wsl][l16][ka + i];
      pf.v[8 + i] = (__bf16)pbuf[wsl][l16][ka + 8 + i];
    }
    const __bf16* avbase = avT + (size_t)((b * H_ + h) * HD_) * 32;
#pragma unroll
    for (int j = 0; j < 8; ++j) {
      FragH vf;
      const uint4* pv = (const uint4*)(avbase + (size_t)(j * 16 + l16) * 32 + kb);
      vf.u[0] = pv[0];
      vf.u[1] = pv[1];
      acc[j] = wmma_bf16(pf, vf, acc[j]);
    }
  }

#pragma unroll
  for (int j = 0; j < 8; ++j) {
#pragma unroll
    for (int r = 0; r < 8; ++r) {
      out[((size_t)(b * S_ + q0 + rb + r)) * D_ + h * HD_ + j * 16 + l16] =
          (__bf16)acc[j][r];
    }
  }
}

// ---------------------------------------------------------------------------
extern "C" void kernel_launch(void* const* d_in, const int* in_sizes, int n_in,
                              void* d_out, int out_size, void* d_ws, size_t ws_size,
                              hipStream_t stream) {
  (void)in_sizes; (void)n_in; (void)out_size; (void)ws_size;
  const float* x       = (const float*)d_in[0];
  const float* adapter = (const float*)d_in[4];
  const float* wq = (const float*)d_in[5];
  const float* wk = (const float*)d_in[6];
  const float* wv = (const float*)d_in[7];
  const float* wo = (const float*)d_in[8];
  const float* lq1 = (const float*)d_in[9];
  const float* lq2 = (const float*)d_in[10];
  const float* lk1 = (const float*)d_in[11];
  const float* lk2 = (const float*)d_in[12];
  const float* lv1 = (const float*)d_in[13];
  const float* lv2 = (const float*)d_in[14];
  const float* lo1 = (const float*)d_in[15];
  const float* lo2 = (const float*)d_in[16];
  const float* gate  = (const float*)d_in[17];
  const float* ngate = (const float*)d_in[18];
  float* out = (float*)d_out;

  const size_t NBSD = (size_t)B_ * S_ * D_;
  const size_t NDD  = (size_t)D_ * D_;
  const size_t NAD  = (size_t)B_ * AL_ * D_;
  const int    Mx   = B_ * S_;  // 4096

  char* p = (char*)d_ws;
  auto take = [&](size_t bytes) -> char* {
    char* r = p;
    p += (bytes + 255) & ~(size_t)255;
    return r;
  };
  __bf16* x_bf    = (__bf16*)take(NBSD * 2);
  __bf16* wq_bf   = (__bf16*)take(NDD * 2);
  __bf16* wk_bf   = (__bf16*)take(NDD * 2);
  __bf16* wv_bf   = (__bf16*)take(NDD * 2);
  __bf16* wo_bf   = (__bf16*)take(NDD * 2);
  __bf16* xq_bf   = (__bf16*)take(NBSD * 2);
  __bf16* xk_bf   = (__bf16*)take(NBSD * 2);
  __bf16* tmp_bf  = (__bf16*)take(NBSD * 2);  // xv, then attention out
  __bf16* vT_bf   = (__bf16*)take(NBSD * 2);
  __bf16* ad_bf   = (__bf16*)take(NAD * 2);
  __bf16* ak_bf   = (__bf16*)take(NAD * 2);
  __bf16* av_bf   = (__bf16*)take(NAD * 2);
  __bf16* avT_bf  = (__bf16*)take((size_t)B_ * H_ * HD_ * 32 * 2);
  __bf16* l1_bf   = (__bf16*)take((size_t)R_ * D_ * 2);
  float*  t_f32   = (float*)take((size_t)Mx * R_ * 4);

  auto cvt = [&](const float* src, __bf16* dst, size_t n) {
    cvt_f32_bf16<<<(unsigned)((n + 255) / 256), 256, 0, stream>>>(src, dst, n);
  };
  // big GEMMs: cooperative LDS-staged kernel (M,N multiples of 128)
  auto gemmc = [&](const __bf16* A, const __bf16* W, float* oF, __bf16* oB,
                   const float* t, const float* l2, int M, int N, int K) {
    int nblk = N / 128;
    gemm_bf16_coop<<<(M / 128) * nblk, 256, 0, stream>>>(A, W, oF, oB, t, l2,
                                                         M, N, K, nblk);
  };
  // small GEMMs: per-wave kernel
  auto gemms = [&](const __bf16* A, const __bf16* W, float* oF, __bf16* oB,
                   int M, int N, int K) {
    int waves  = ((M + 15) / 16) * ((N + 127) / 128);
    int blocks = (waves + 7) / 8;
    gemm_bf16_wmma<<<blocks, 256, 0, stream>>>(A, W, oF, oB, M, N, K);
  };

  // 1) convert inputs
  cvt(x, x_bf, NBSD);
  cvt(wq, wq_bf, NDD);
  cvt(wk, wk_bf, NDD);
  cvt(wv, wv_bf, NDD);
  cvt(wo, wo_bf, NDD);
  cvt(adapter, ad_bf, NAD);

  // 2) xq = x@wq^T + (x@lq1^T)@lq2^T
  cvt(lq1, l1_bf, (size_t)R_ * D_);
  gemms(x_bf, l1_bf, t_f32, nullptr, Mx, R_, D_);
  gemmc(x_bf, wq_bf, nullptr, xq_bf, t_f32, lq2, Mx, D_, D_);

  // 3) xk
  cvt(lk1, l1_bf, (size_t)R_ * D_);
  gemms(x_bf, l1_bf, t_f32, nullptr, Mx, R_, D_);
  gemmc(x_bf, wk_bf, nullptr, xk_bf, t_f32, lk2, Mx, D_, D_);

  // 4) xv -> vT
  cvt(lv1, l1_bf, (size_t)R_ * D_);
  gemms(x_bf, l1_bf, t_f32, nullptr, Mx, R_, D_);
  gemmc(x_bf, wv_bf, nullptr, tmp_bf, t_f32, lv2, Mx, D_, D_);
  transpose_v_kernel<<<(unsigned)((NBSD + 255) / 256), 256, 0, stream>>>(tmp_bf,
                                                                         vT_bf);

  // 5) adapter K/V (no LoRA on adapter path)
  gemms(ad_bf, wk_bf, nullptr, ak_bf, B_ * AL_, D_, D_);
  gemms(ad_bf, wv_bf, nullptr, av_bf, B_ * AL_, D_, D_);
  {
    int n = B_ * H_ * HD_ * 32;
    make_avT_kernel<<<(n + 255) / 256, 256, 0, stream>>>(av_bf, avT_bf);
  }

  // 6) attention (main flash + adapter), output into tmp_bf
  flash_attn_wmma<<<(B_ * H_ * (S_ / 16)) / 8, 256, 0, stream>>>(
      xq_bf, xk_bf, vT_bf, ak_bf, avT_bf, gate, ngate, tmp_bf);

  // 7) y = attn@wo^T + (attn@lo1^T)@lo2^T  (f32 to d_out)
  cvt(lo1, l1_bf, (size_t)R_ * D_);
  gemms(tmp_bf, l1_bf, t_f32, nullptr, Mx, R_, D_);
  gemmc(tmp_bf, wo_bf, out, nullptr, t_f32, lo2, Mx, D_, D_);
}
